// PointnetFPModule_7808250544755
// MI455X (gfx1250) — matrix operand order repaired and checked by reading, not compiled, not run on hardware
//
#include <hip/hip_runtime.h>
#include <hip/hip_bf16.h>

#define BATCH 4
#define NPTS  16384
#define MPTS  4096
#define C1    256
#define C2    512
#define CIN   768
#define CH    256
#define COLS  (BATCH * NPTS)   // 65536

typedef unsigned short u16;
typedef unsigned int   u32;

typedef __attribute__((ext_vector_type(16))) __bf16 v16bf;
typedef __attribute__((ext_vector_type(8)))  float  v8f;
typedef int v4i __attribute__((vector_size(16)));   // matches builtin param type

union Frag { v16bf v; uint4 u[2]; };

__device__ __forceinline__ u16 f2bf(float f) {
    u32 u = __float_as_uint(f);
    u += 0x7fffu + ((u >> 16) & 1u);   // round-to-nearest-even
    return (u16)(u >> 16);
}
__device__ __forceinline__ float bf2f(u16 h) {
    return __uint_as_float(((u32)h) << 16);
}

// ---------------------------------------------------------------------------
// Async global->LDS copy (CDNA5 GLOBAL_LOAD_ASYNC_TO_LDS_B128, ASYNCcnt),
// with a synchronous fallback if the builtin is absent on this toolchain.
// ---------------------------------------------------------------------------
#if __has_builtin(__builtin_amdgcn_global_load_async_to_lds_b128)
#define ASYNC_LDS 1
#else
#define ASYNC_LDS 0
#endif

#define AS_GLOBAL __attribute__((address_space(1)))
#define AS_LDS    __attribute__((address_space(3)))

__device__ __forceinline__ void cp_b128(const u16* g, u16* l) {
#if ASYNC_LDS
    __builtin_amdgcn_global_load_async_to_lds_b128(
        (AS_GLOBAL v4i*)(AS_GLOBAL void*)(void*)(u16*)g,
        (AS_LDS v4i*)(AS_LDS void*)l, 0, 0);
#else
    *(uint4*)l = *(const uint4*)g;
#endif
}

__device__ __forceinline__ void wait_async() {
#if ASYNC_LDS
 #if __has_builtin(__builtin_amdgcn_s_wait_asynccnt)
    __builtin_amdgcn_s_wait_asynccnt(0);
 #else
    asm volatile("s_wait_asynccnt 0x0" ::: "memory");
 #endif
#endif
}

// ---------------------------------------------------------------------------
// K0: brute-force 3-NN per query point; known points staged via LDS.
// ---------------------------------------------------------------------------
__global__ __launch_bounds__(256) void knn3_kernel(
    const float* __restrict__ unknown, const float* __restrict__ known,
    int* __restrict__ idx, float* __restrict__ wgt)
{
    const int b   = blockIdx.y;
    const int n   = blockIdx.x * 256 + threadIdx.x;
    const int col = b * NPTS + n;
    const float ux = unknown[col * 3 + 0];
    const float uy = unknown[col * 3 + 1];
    const float uz = unknown[col * 3 + 2];

    __shared__ float kx[512], ky[512], kz[512];

    float d0 = 3.4e38f, d1 = 3.4e38f, d2 = 3.4e38f;
    int   j0 = 0, j1 = 0, j2 = 0;

    for (int c0 = 0; c0 < MPTS; c0 += 512) {
        __syncthreads();
        for (int j = threadIdx.x; j < 512; j += 256) {
            const float* p = known + ((size_t)b * MPTS + c0 + j) * 3;
            kx[j] = p[0]; ky[j] = p[1]; kz[j] = p[2];
        }
        __syncthreads();
        for (int j = 0; j < 512; ++j) {
            const float dx = ux - kx[j];
            const float dy = uy - ky[j];
            const float dz = uz - kz[j];
            const float d  = fmaf(dx, dx, fmaf(dy, dy, dz * dz));
            const int  gj  = c0 + j;
            if (d < d0)      { d2 = d1; j2 = j1; d1 = d0; j1 = j0; d0 = d; j0 = gj; }
            else if (d < d1) { d2 = d1; j2 = j1; d1 = d;  j1 = gj; }
            else if (d < d2) { d2 = d;  j2 = gj; }
        }
    }
    const float r0 = 1.0f / (d0 + 1e-8f);
    const float r1 = 1.0f / (d1 + 1e-8f);
    const float r2 = 1.0f / (d2 + 1e-8f);
    const float rs = 1.0f / (r0 + r1 + r2);
    idx[col * 3 + 0] = j0; idx[col * 3 + 1] = j1; idx[col * 3 + 2] = j2;
    wgt[col * 3 + 0] = r0 * rs; wgt[col * 3 + 1] = r1 * rs; wgt[col * 3 + 2] = r2 * rs;
}

// ---------------------------------------------------------------------------
// K1: three_interpolate + concat -> Xbf column-major [col][768] bf16
// ---------------------------------------------------------------------------
__global__ __launch_bounds__(256) void interp_concat_kernel(
    const int* __restrict__ idx, const float* __restrict__ wgt,
    const float* __restrict__ known_feats, const float* __restrict__ unknow_feats,
    u16* __restrict__ Xbf)
{
    const int col = blockIdx.x * 256 + threadIdx.x;
    const int b   = col >> 14;
    const int n   = col & (NPTS - 1);
    const int i0 = idx[col * 3 + 0], i1 = idx[col * 3 + 1], i2 = idx[col * 3 + 2];
    const float w0 = wgt[col * 3 + 0], w1 = wgt[col * 3 + 1], w2 = wgt[col * 3 + 2];

    const float* kf = known_feats + (size_t)b * C2 * MPTS;
    u16* dst = Xbf + (size_t)col * CIN;

    union { u16 h[8]; uint4 q; } pk;
    for (int c0 = 0; c0 < C2; c0 += 8) {
        for (int r = 0; r < 8; ++r) {
            const float* kc = kf + (size_t)(c0 + r) * MPTS;
            pk.h[r] = f2bf(w0 * kc[i0] + w1 * kc[i1] + w2 * kc[i2]);
        }
        *(uint4*)(dst + c0) = pk.q;
    }
    const float* uf = unknow_feats + (size_t)b * C1 * NPTS + n;
    for (int c0 = 0; c0 < C1; c0 += 8) {
        for (int r = 0; r < 8; ++r) pk.h[r] = f2bf(uf[(size_t)(c0 + r) * NPTS]);
        *(uint4*)(dst + C2 + c0) = pk.q;
    }
}

// ---------------------------------------------------------------------------
// K2: convert W1 (256x768) and W2 (256x256) f32 -> bf16
// ---------------------------------------------------------------------------
__global__ __launch_bounds__(256) void cvt_w_kernel(
    const float* __restrict__ W1, const float* __restrict__ W2,
    u16* __restrict__ W1bf, u16* __restrict__ W2bf)
{
    const int i = blockIdx.x * 256 + threadIdx.x;
    if (i < CH * CIN) W1bf[i] = f2bf(W1[i]);
    const int j = i - CH * CIN;
    if (j >= 0 && j < CH * CH) W2bf[j] = f2bf(W2[j]);
}

// ---------------------------------------------------------------------------
// GEMM: y[256 x COLS] = A[256 x CK] * B[CK x COLS] + bias
// A row-major bf16; B column-major bf16 ([col][CK]).
// Block: 256 thr = 8 wave32, tile 64(o) x 128(col); wave tile 16x64.
// Double-buffered LDS; async global->LDS copies overlap the next K-step's
// fetch with the current step's WMMAs.
// STORE_F32==0: bf16 out [col][256].  STORE_F32==1: f32 out [b][c][n].
// ---------------------------------------------------------------------------
template<int CK, int STORE_F32>
__global__ __launch_bounds__(256) void gemm_kernel(
    const u16* __restrict__ A, const u16* __restrict__ Bm,
    const float* __restrict__ bias, u16* __restrict__ outBf,
    float* __restrict__ outF)
{
    __shared__ u16 As[2][64 * 32];    // [buf][row][k]
    __shared__ u16 Bs[2][128 * 32];   // [buf][col][k]

    const int tid  = threadIdx.x;
    const int lane = tid & 31;
    const int wid  = tid >> 5;
    const int wm   = wid & 3;       // o sub-tile
    const int wn   = wid >> 2;      // col sub-tile (0..1)
    const int blockM   = blockIdx.y * 64;
    const int blockCol = blockIdx.x * 128;

    v8f acc[4];
    for (int t = 0; t < 4; ++t)
        for (int e = 0; e < 8; ++e) acc[t][e] = 0.0f;

    const int arowLd = tid >> 2, achLd = tid & 3;   // A tile: 64 rows x 4x16B
    const int bcolLd = tid >> 1, bchLd = tid & 1;   // B tile: 128 cols x 2x32B
    const int laneHi = (lane & 16) ? 1 : 0;
    const int arow   = wm * 16 + (lane & 15);

    const u16* aSrcBase = A  + (size_t)(blockM   + arowLd) * CK + achLd * 8;
    const u16* bSrcBase = Bm + (size_t)(blockCol + bcolLd) * CK + bchLd * 16;

    auto load_tile = [&](int k0, int buf) {
        cp_b128(aSrcBase + k0, &As[buf][arowLd * 32 + achLd * 8]);
        u16* bDst = &Bs[buf][bcolLd * 32 + bchLd * 16];
        cp_b128(bSrcBase + k0,     bDst);
        cp_b128(bSrcBase + k0 + 8, bDst + 8);
    };

    constexpr int NSTEP = CK / 32;

    load_tile(0, 0);
    wait_async();
    __syncthreads();

    for (int s = 0; s < NSTEP; ++s) {
        const int cur = s & 1;
        if (s + 1 < NSTEP) load_tile((s + 1) * 32, cur ^ 1);

        // A fragment: M = lane&15, K runs {0-7,16-23} (lanes<16) / {8-15,24-31}
        Frag a;
        const uint4* ap = (const uint4*)(&As[cur][arow * 32] + laneHi * 8);
        a.u[0] = ap[0];
        a.u[1] = ap[2];

        // Load all 4 B fragments first, then issue WMMAs back-to-back.
        Frag bfr[4];
        for (int t = 0; t < 4; ++t) {
            const int bcol = wn * 64 + t * 16 + (lane & 15);
            const uint4* bp = (const uint4*)(&Bs[cur][bcol * 32] + laneHi * 8);
            bfr[t].u[0] = bp[0];
            bfr[t].u[1] = bp[2];
        }
        for (int t = 0; t < 4; ++t) {
            acc[t] = __builtin_amdgcn_wmma_f32_16x16x32_bf16(
                false, a.v, false, bfr[t].v, (short)0, acc[t], false, false);
        }

        wait_async();
        __syncthreads();
    }

    // C/D layout: VGPR r -> M = (lane<16 ? r : r+8), N = lane&15
    const int obase = blockM + wm * 16 + laneHi * 8;
    float bv[8];
    for (int r = 0; r < 8; ++r) bv[r] = bias[obase + r];

    for (int t = 0; t < 4; ++t) {
        const int col = blockCol + wn * 64 + t * 16 + (lane & 15);
        if (STORE_F32) {
            const int bb = col >> 14;
            const int nn = col & (NPTS - 1);
            float* o = outF + ((size_t)(bb * CH + obase)) * NPTS + nn;
            for (int r = 0; r < 8; ++r) o[(size_t)r * NPTS] = acc[t][r] + bv[r];
        } else {
            union { u16 h[8]; uint4 q; } pk;
            for (int r = 0; r < 8; ++r) pk.h[r] = f2bf(acc[t][r] + bv[r]);
            *(uint4*)(outBf + (size_t)col * CH + obase) = pk.q;
        }
    }
}

// ---------------------------------------------------------------------------
// BN statistics (sum, sumsq per channel) — deterministic block reductions
// ---------------------------------------------------------------------------
__global__ __launch_bounds__(256) void stats_bf_kernel(
    const u16* __restrict__ y, float* __restrict__ stats)
{
    const int c = blockIdx.x;
    float s = 0.f, q = 0.f;
    for (int col = threadIdx.x; col < COLS; col += 256) {
        const float v = bf2f(y[(size_t)col * CH + c]);
        s += v; q += v * v;
    }
    __shared__ float rs[256], rq[256];
    rs[threadIdx.x] = s; rq[threadIdx.x] = q;
    __syncthreads();
    for (int o = 128; o > 0; o >>= 1) {
        if (threadIdx.x < o) { rs[threadIdx.x] += rs[threadIdx.x + o];
                               rq[threadIdx.x] += rq[threadIdx.x + o]; }
        __syncthreads();
    }
    if (threadIdx.x == 0) { stats[c] = rs[0]; stats[CH + c] = rq[0]; }
}

__global__ __launch_bounds__(256) void stats_f32_kernel(
    const float* __restrict__ y, float* __restrict__ stats)
{
    const int c = blockIdx.x;
    float s = 0.f, q = 0.f;
    for (int b = 0; b < BATCH; ++b) {
        const float* p = y + ((size_t)(b * CH + c)) * NPTS;
        for (int n = threadIdx.x; n < NPTS; n += 256) {
            const float v = p[n];
            s += v; q += v * v;
        }
    }
    __shared__ float rs[256], rq[256];
    rs[threadIdx.x] = s; rq[threadIdx.x] = q;
    __syncthreads();
    for (int o = 128; o > 0; o >>= 1) {
        if (threadIdx.x < o) { rs[threadIdx.x] += rs[threadIdx.x + o];
                               rq[threadIdx.x] += rq[threadIdx.x + o]; }
        __syncthreads();
    }
    if (threadIdx.x == 0) { stats[c] = rs[0]; stats[CH + c] = rq[0]; }
}

// ---------------------------------------------------------------------------
// BN + ReLU
// ---------------------------------------------------------------------------
__global__ __launch_bounds__(256) void bnrelu_bf_kernel(
    const u16* __restrict__ y, const float* __restrict__ stats,
    const float* __restrict__ g, const float* __restrict__ bt,
    u16* __restrict__ h)
{
    const u32 i = blockIdx.x * 256 + threadIdx.x;   // [col][256] layout
    const int c = i & (CH - 1);
    const float inv  = 1.0f / (float)COLS;
    const float mean = stats[c] * inv;
    const float var  = stats[CH + c] * inv - mean * mean;
    const float v    = bf2f(y[i]);
    float o = (v - mean) * rsqrtf(var + 1e-5f) * g[c] + bt[c];
    h[i] = f2bf(o > 0.f ? o : 0.f);
}

__global__ __launch_bounds__(256) void bnrelu_f32_kernel(
    float* __restrict__ y, const float* __restrict__ stats,
    const float* __restrict__ g, const float* __restrict__ bt)
{
    const u32 i = blockIdx.x * 256 + threadIdx.x;   // [b][c][n] layout
    const int c = (i >> 14) & (CH - 1);
    const float inv  = 1.0f / (float)COLS;
    const float mean = stats[c] * inv;
    const float var  = stats[CH + c] * inv - mean * mean;
    const float v    = y[i];
    float o = (v - mean) * rsqrtf(var + 1e-5f) * g[c] + bt[c];
    y[i] = o > 0.f ? o : 0.f;
}

// ---------------------------------------------------------------------------
extern "C" void kernel_launch(void* const* d_in, const int* in_sizes, int n_in,
                              void* d_out, int out_size, void* d_ws, size_t ws_size,
                              hipStream_t stream)
{
    const float* unknown      = (const float*)d_in[0];
    const float* known        = (const float*)d_in[1];
    const float* unknow_feats = (const float*)d_in[2];
    const float* known_feats  = (const float*)d_in[3];
    const float* W1  = (const float*)d_in[4];
    const float* b1  = (const float*)d_in[5];
    const float* g1  = (const float*)d_in[6];
    const float* bt1 = (const float*)d_in[7];
    const float* W2  = (const float*)d_in[8];
    const float* b2  = (const float*)d_in[9];
    const float* g2  = (const float*)d_in[10];
    const float* bt2 = (const float*)d_in[11];
    float* out = (float*)d_out;

    char* ws = (char*)d_ws;
    int*   idx    = (int*)  (ws + 0);           // 786,432 B
    float* wgt    = (float*)(ws + 786432);      // 786,432 B
    u16*   W1bf   = (u16*)  (ws + 1572864);     // 393,216 B
    u16*   W2bf   = (u16*)  (ws + 1966080);     // 131,072 B
    float* stats1 = (float*)(ws + 2097152);     // 2,048 B
    float* stats2 = (float*)(ws + 2099200);     // 2,048 B
    u16*   Xbf    = (u16*)  (ws + 4194304);     // 100,663,296 B [col][768]
    u16*   y1bf   = (u16*)  (ws + 104857600);   // 33,554,432 B [col][256]
    u16*   h1bf   = (u16*)  (ws + 138412032);   // 33,554,432 B [col][256]

    knn3_kernel<<<dim3(NPTS / 256, BATCH), 256, 0, stream>>>(unknown, known, idx, wgt);
    interp_concat_kernel<<<COLS / 256, 256, 0, stream>>>(idx, wgt, known_feats,
                                                         unknow_feats, Xbf);
    cvt_w_kernel<<<(CH * CIN + CH * CH) / 256, 256, 0, stream>>>(W1, W2, W1bf, W2bf);

    gemm_kernel<CIN, 0><<<dim3(COLS / 128, CH / 64), 256, 0, stream>>>(
        W1bf, Xbf, b1, y1bf, nullptr);
    stats_bf_kernel<<<CH, 256, 0, stream>>>(y1bf, stats1);
    bnrelu_bf_kernel<<<(CH * COLS) / 256, 256, 0, stream>>>(y1bf, stats1, g1, bt1, h1bf);

    gemm_kernel<CH, 1><<<dim3(COLS / 128, CH / 64), 256, 0, stream>>>(
        W2bf, h1bf, b2, nullptr, out);
    stats_f32_kernel<<<CH, 256, 0, stream>>>(out, stats2);
    bnrelu_f32_kernel<<<(CH * COLS) / 256, 256, 0, stream>>>(out, stats2, g2, bt2);
}
